// MultiHeadAttention_90168543412843
// MI455X (gfx1250) — matrix operand order repaired and checked by reading, not compiled, not run on hardware
//
#include <hip/hip_runtime.h>
#include <hip/hip_bf16.h>

typedef __attribute__((ext_vector_type(16))) _Float16 v16h;
typedef __attribute__((ext_vector_type(8)))  _Float16 v8h;
typedef __attribute__((ext_vector_type(8)))  float    v8f;
typedef __attribute__((ext_vector_type(4)))  float    v4f;

static constexpr int kS  = 2048;
static constexpr int kD  = 768;
static constexpr int kH  = 12;
static constexpr int kDk = 64;
static constexpr size_t kPerTensor = (size_t)2 * kS * kD;   // 3,145,728 elems
static constexpr size_t kPerWeight = (size_t)kD * kD;       //   589,824 elems

#define WMMA_F16(A, B, C) \
  __builtin_amdgcn_wmma_f32_16x16x32_f16(false, (A), false, (B), (short)0, (C), false, false)

// ---------------------------------------------------------------------------
// Fragment loaders (layouts per CDNA5 ISA 7.12.2, wave32)
// A 16x32 f16: lane<16 -> row M=lane, K {0..7,16..23}; lane>=16 -> M=lane-16, K {8..15,24..31}
// B 32x16 f16: lane<16 -> col N=lane, K {0..7,16..23}; lane>=16 -> N=lane-16, K {8..15,24..31}
// C/D 16x16 f32: lane -> N=lane&15; VGPR r -> M = r + 8*(lane>=16)
// ---------------------------------------------------------------------------
__device__ __forceinline__ v16h a_frag_f16(const _Float16* base, int ld,
                                           int row0, int k0, int lane) {
  const int m   = lane & 15;
  const int off = (lane & 16) ? 8 : 0;
  const _Float16* p = base + (size_t)(row0 + m) * ld + k0 + off;
  v8h lo = *(const v8h*)p;
  v8h hi = *(const v8h*)(p + 16);
  v16h a;
#pragma unroll
  for (int i = 0; i < 8; ++i) { a[i] = lo[i]; a[i + 8] = hi[i]; }
  return a;
}

__device__ __forceinline__ v16h b_frag_f16(const _Float16* __restrict__ base, int ld,
                                           int col0, int k0, int lane) {
  const int j  = lane & 15;
  const int kb = k0 + ((lane & 16) ? 8 : 0);
  const _Float16* p = base + (size_t)(col0 + j) * ld + kb;
  v8h lo = *(const v8h*)p;
  v8h hi = *(const v8h*)(p + 16);
  v16h b;
#pragma unroll
  for (int i = 0; i < 8; ++i) { b[i] = lo[i]; b[i + 8] = hi[i]; }
  return b;
}

__device__ __forceinline__ float half_reduce_max(float v) {
  v = fmaxf(v, __shfl_xor(v, 1));
  v = fmaxf(v, __shfl_xor(v, 2));
  v = fmaxf(v, __shfl_xor(v, 4));
  v = fmaxf(v, __shfl_xor(v, 8));
  return v;
}
__device__ __forceinline__ float half_reduce_sum(float v) {
  v += __shfl_xor(v, 1);
  v += __shfl_xor(v, 2);
  v += __shfl_xor(v, 4);
  v += __shfl_xor(v, 8);
  return v;
}

// ---------------------------------------------------------------------------
// Kernel 0: one-shot f32 -> f16 conversion (8 elems/thread, b128 in, b128 out).
// ---------------------------------------------------------------------------
__global__ void __launch_bounds__(256)
cvt_f16_kernel(const float* __restrict__ src, _Float16* __restrict__ dst, int n8) {
  const int i = blockIdx.x * 256 + threadIdx.x;
  if (i >= n8) return;
  const v4f* s = (const v4f*)src + (size_t)i * 2;
  v4f x0 = s[0], x1 = s[1];
  v8h d;
#pragma unroll
  for (int j = 0; j < 4; ++j) { d[j] = (_Float16)x0[j]; d[j + 4] = (_Float16)x1[j]; }
  ((v8h*)dst)[i] = d;
}

// ---------------------------------------------------------------------------
// Kernel 1: Lin = X @ W^T + b, all-f16 operands, f32 accumulate.
// Wave tile: 32 rows x 64 cols (8 wmma / k-step), double-buffered fragments.
// Plain store == head layout (B,H,S,64) for Q/K (raw-view equivalence).
// writeVt: store V transposed per head as (B,H,64,S).
// ---------------------------------------------------------------------------
__global__ void __launch_bounds__(128)
proj_kernel(const _Float16* __restrict__ Xh, const _Float16* __restrict__ Wh,
            const float* __restrict__ bias,
            _Float16* __restrict__ outPlain, _Float16* __restrict__ outVt,
            int writeVt) {
  const int lane = threadIdx.x & 31;
  const int wave = threadIdx.x >> 5;
  const int row0 = blockIdx.x * 128 + wave * 32;  // rows of X (n = b*S + s')
  const int col0 = blockIdx.y * 64;               // output columns m

  v8f acc[2][4] = {};

  // Prime the pipeline: k-tile 0.
  v16h a0 = a_frag_f16(Xh, kD, row0, 0, lane);
  v16h a1 = a_frag_f16(Xh, kD, row0 + 16, 0, lane);
  v16h bw0 = b_frag_f16(Wh, kD, col0,      0, lane);
  v16h bw1 = b_frag_f16(Wh, kD, col0 + 16, 0, lane);
  v16h bw2 = b_frag_f16(Wh, kD, col0 + 32, 0, lane);
  v16h bw3 = b_frag_f16(Wh, kD, col0 + 48, 0, lane);

  for (int k0 = 0; k0 < kD; k0 += 32) {
    // Preload next k-tile into fresh registers (wraps harmlessly on last iter).
    const int kn = (k0 + 32 < kD) ? (k0 + 32) : 0;
    v16h na0 = a_frag_f16(Xh, kD, row0, kn, lane);
    v16h na1 = a_frag_f16(Xh, kD, row0 + 16, kn, lane);
    v16h nb0 = b_frag_f16(Wh, kD, col0,      kn, lane);
    v16h nb1 = b_frag_f16(Wh, kD, col0 + 16, kn, lane);
    v16h nb2 = b_frag_f16(Wh, kD, col0 + 32, kn, lane);
    v16h nb3 = b_frag_f16(Wh, kD, col0 + 48, kn, lane);

    acc[0][0] = WMMA_F16(a0, bw0, acc[0][0]);
    acc[1][0] = WMMA_F16(a1, bw0, acc[1][0]);
    acc[0][1] = WMMA_F16(a0, bw1, acc[0][1]);
    acc[1][1] = WMMA_F16(a1, bw1, acc[1][1]);
    acc[0][2] = WMMA_F16(a0, bw2, acc[0][2]);
    acc[1][2] = WMMA_F16(a1, bw2, acc[1][2]);
    acc[0][3] = WMMA_F16(a0, bw3, acc[0][3]);
    acc[1][3] = WMMA_F16(a1, bw3, acc[1][3]);

    a0 = na0; a1 = na1;
    bw0 = nb0; bw1 = nb1; bw2 = nb2; bw3 = nb3;
  }

  const int nj   = lane & 15;
  const int moff = (lane & 16) ? 8 : 0;
#pragma unroll
  for (int rt = 0; rt < 2; ++rt) {
#pragma unroll
    for (int g = 0; g < 4; ++g) {
      const int m  = col0 + g * 16 + nj;
      const float bv = bias[m];
#pragma unroll
      for (int r = 0; r < 8; ++r) {
        const float val = acc[rt][g][r] + bv;
        const int n = row0 + rt * 16 + r + moff;  // global row in [0, 4096)
        if (!writeVt) {
          outPlain[(size_t)n * kD + m] = (_Float16)val;
        } else {
          const int b  = n >> 11;            // n / 2048
          const int sp = n & 2047;
          const int flat = sp * kD + m;      // raw-view flat index within batch
          const int h = flat >> 17;          // / (S*dk)
          const int s = (flat >> 6) & 2047;
          const int d = flat & 63;
          outVt[(((size_t)b * kH + h) * kDk + d) * kS + s] = (_Float16)val;
        }
      }
    }
  }
}

// ---------------------------------------------------------------------------
// Kernel 2: flash attention. 128 threads = 4 waves; wave owns a 16-query tile,
// loops over keys 32-wide. Per step: 4 wmma (QK^T), V B-frags issued early,
// mask(-1e9) + online softmax (shfl butterflies + v_exp), P transposed to
// A-layout via wave-private LDS, 4 wmma (P@V) into 16x64 f32 accumulators.
// ---------------------------------------------------------------------------
__global__ void __launch_bounds__(128)
attn_kernel(const _Float16* __restrict__ Qh, const _Float16* __restrict__ Kh,
            const _Float16* __restrict__ Vt, const int* __restrict__ mask,
            float* __restrict__ out) {
  __shared__ __align__(64) _Float16 Plds[4][16 * 32];

  const int lane = threadIdx.x & 31;
  const int wave = threadIdx.x >> 5;
  const int q0 = blockIdx.x * 64 + wave * 16;
  const int h  = blockIdx.y;
  const int b  = blockIdx.z;

  const _Float16* Qbh = Qh + ((size_t)b * kH + h) * (kS * kDk);
  const _Float16* Kbh = Kh + ((size_t)b * kH + h) * (kS * kDk);
  const _Float16* Vbh = Vt + ((size_t)b * kH + h) * (kDk * kS);
  const int* maskb    = mask + (size_t)b * kS * kS;

  const v16h aq0 = a_frag_f16(Qbh, kDk, q0, 0, lane);
  const v16h aq1 = a_frag_f16(Qbh, kDk, q0, 32, lane);

  v8f O[4] = {};
  float mrow[8], lrow[8];
#pragma unroll
  for (int r = 0; r < 8; ++r) { mrow[r] = -3.0e38f; lrow[r] = 0.0f; }

  const int   nj    = lane & 15;
  const int   moff  = (lane & 16) ? 8 : 0;
  const float scale = 0.125f;  // 1/sqrt(64)
  _Float16* pl = Plds[wave];   // wave-private: no inter-wave barrier needed

  for (int k0 = 0; k0 < kS; k0 += 32) {
    if (k0 + 32 < kS) {  // uniform -> global_prefetch_b8
      __builtin_prefetch(Kbh + (size_t)(k0 + 32) * kDk, 0, 0);
      __builtin_prefetch(maskb + (size_t)q0 * kS + k0 + 32, 0, 0);
    }

    // ---- scores: two 16x16 tiles, dk split 0..31 / 32..63 ----
    v8f sc[2];
#pragma unroll
    for (int t = 0; t < 2; ++t) {
      v16h bk0 = b_frag_f16(Kbh, kDk, k0 + t * 16, 0, lane);
      v16h bk1 = b_frag_f16(Kbh, kDk, k0 + t * 16, 32, lane);
      v8f z = {};
      z = WMMA_F16(aq0, bk0, z);
      sc[t] = WMMA_F16(aq1, bk1, z);
    }

    // Issue V B-fragment loads now; latency overlaps the softmax below.
    v16h bv0 = b_frag_f16(Vbh, kS, 0,  k0, lane);
    v16h bv1 = b_frag_f16(Vbh, kS, 16, k0, lane);
    v16h bv2 = b_frag_f16(Vbh, kS, 32, k0, lane);
    v16h bv3 = b_frag_f16(Vbh, kS, 48, k0, lane);

    // ---- scale then mask (masked positions get exactly -1e9, as reference) ----
#pragma unroll
    for (int t = 0; t < 2; ++t) {
#pragma unroll
      for (int r = 0; r < 8; ++r) {
        const int qrow = q0 + r + moff;
        const int kcol = k0 + t * 16 + nj;
        const int mk = maskb[(size_t)qrow * kS + kcol];
        const float s = sc[t][r] * scale;
        sc[t][r] = (mk == 0) ? -1.0e9f : s;
      }
    }

    // ---- online softmax ----
    float corr[8], mnew[8];
#pragma unroll
    for (int r = 0; r < 8; ++r) {
      float v = fmaxf(sc[0][r], sc[1][r]);
      v = half_reduce_max(v);
      mnew[r] = fmaxf(mrow[r], v);
      corr[r] = __expf(mrow[r] - mnew[r]);
      mrow[r] = mnew[r];
    }
#pragma unroll
    for (int r = 0; r < 8; ++r) {
      const float p0 = __expf(sc[0][r] - mnew[r]);
      const float p1 = __expf(sc[1][r] - mnew[r]);
      sc[0][r] = p0;
      sc[1][r] = p1;
      lrow[r] = lrow[r] * corr[r] + half_reduce_sum(p0 + p1);
    }
#pragma unroll
    for (int c = 0; c < 4; ++c)
#pragma unroll
      for (int r = 0; r < 8; ++r) O[c][r] *= corr[r];

    // ---- C-layout -> A-layout transpose of P through wave-private LDS ----
#pragma unroll
    for (int t = 0; t < 2; ++t)
#pragma unroll
      for (int r = 0; r < 8; ++r)
        pl[(r + moff) * 32 + t * 16 + nj] = (_Float16)sc[t][r];
    // Same-wave DS ops are in-order; compiler tracks the VGPR dependency.
    v16h ap = a_frag_f16(pl, 32, 0, 0, lane);

    // ---- O += P @ V ----
    O[0] = WMMA_F16(ap, bv0, O[0]);
    O[1] = WMMA_F16(ap, bv1, O[1]);
    O[2] = WMMA_F16(ap, bv2, O[2]);
    O[3] = WMMA_F16(ap, bv3, O[3]);
  }

  // ---- normalize and store (B,H,S,64) f32 ----
  float* outbh = out + ((size_t)b * kH + h) * (kS * kDk);
#pragma unroll
  for (int r = 0; r < 8; ++r) {
    const float inv = 1.0f / lrow[r];
    const int qrow = q0 + r + moff;
#pragma unroll
    for (int c = 0; c < 4; ++c)
      outbh[(size_t)qrow * kDk + c * 16 + nj] = O[c][r] * inv;
  }
}

// ---------------------------------------------------------------------------
extern "C" void kernel_launch(void* const* d_in, const int* in_sizes, int n_in,
                              void* d_out, int out_size, void* d_ws, size_t ws_size,
                              hipStream_t stream) {
  const float* q    = (const float*)d_in[0];
  const float* k    = (const float*)d_in[1];
  const float* v    = (const float*)d_in[2];
  const int*   mask = (const int*)d_in[3];
  const float* Wq   = (const float*)d_in[4];
  const float* bq   = (const float*)d_in[5];
  const float* Wk   = (const float*)d_in[6];
  const float* bk   = (const float*)d_in[7];
  const float* Wv   = (const float*)d_in[8];
  const float* bv   = (const float*)d_in[9];

  _Float16* Qh  = (_Float16*)d_ws;          // (B,S,D) f16 == (B,H,S,64) raw view
  _Float16* Kh  = Qh  + kPerTensor;
  _Float16* Vt  = Kh  + kPerTensor;         // (B,H,64,S) f16
  _Float16* Xq  = Vt  + kPerTensor;         // f16 copies of inputs
  _Float16* Xk  = Xq  + kPerTensor;
  _Float16* Xv  = Xk  + kPerTensor;
  _Float16* Wqh = Xv  + kPerTensor;         // f16 copies of weights
  _Float16* Wkh = Wqh + kPerWeight;
  _Float16* Wvh = Wkh + kPerWeight;

  // One-shot conversions (each elem touched once; L2-resident afterwards).
  const int nT8 = (int)(kPerTensor / 8), nW8 = (int)(kPerWeight / 8);
  cvt_f16_kernel<<<(nT8 + 255) / 256, 256, 0, stream>>>(q,  Xq,  nT8);
  cvt_f16_kernel<<<(nT8 + 255) / 256, 256, 0, stream>>>(k,  Xk,  nT8);
  cvt_f16_kernel<<<(nT8 + 255) / 256, 256, 0, stream>>>(v,  Xv,  nT8);
  cvt_f16_kernel<<<(nW8 + 255) / 256, 256, 0, stream>>>(Wq, Wqh, nW8);
  cvt_f16_kernel<<<(nW8 + 255) / 256, 256, 0, stream>>>(Wk, Wkh, nW8);
  cvt_f16_kernel<<<(nW8 + 255) / 256, 256, 0, stream>>>(Wv, Wvh, nW8);

  dim3 gproj(32, 12, 1);                    // 4096/128 rows, 768/64 cols
  proj_kernel<<<gproj, 128, 0, stream>>>(Xq, Wqh, bq, Qh, nullptr, 0);
  proj_kernel<<<gproj, 128, 0, stream>>>(Xk, Wkh, bk, Kh, nullptr, 0);
  proj_kernel<<<gproj, 128, 0, stream>>>(Xv, Wvh, bv, nullptr, Vt, 1);

  dim3 gattn(kS / 64, kH, 2);               // 32 x 12 x 2
  attn_kernel<<<gattn, 128, 0, stream>>>(Qh, Kh, Vt, mask, (float*)d_out);
}
